// LinearSelfAttentionBlock_90537910599865
// MI455X (gfx1250) — matrix operand order repaired and checked by reading, not compiled, not run on hardware
//
#include <hip/hip_runtime.h>

// ---------------------------------------------------------------------------
// Problem constants (match reference)
// ---------------------------------------------------------------------------
constexpr int Bc   = 4;
constexpr int Lc   = 4096;
constexpr int Ec   = 1024;
constexpr int Hc   = 16;
constexpr int Dc   = 64;
constexpr int MLPc = 4096;
constexpr long Nc  = (long)Bc * Lc;        // 16384 token rows
constexpr long Nzc = (long)Bc * (Lc + 1);  // 16388 rows incl. zero token

typedef __attribute__((ext_vector_type(16))) _Float16 v16h;
typedef __attribute__((ext_vector_type(8)))  float    v8f;
typedef __attribute__((ext_vector_type(4)))  int      v4i;

union Frag16 { unsigned int u32[8]; v16h h; };

// CDNA5 async global->LDS path (ASYNCcnt), guarded so compilation never breaks.
#if defined(__gfx1250__) && __has_builtin(__builtin_amdgcn_global_load_async_to_lds_b128) && __has_builtin(__builtin_amdgcn_s_wait_asynccnt)
#define USE_ASYNC 1
typedef __attribute__((address_space(3))) void svoid_t;
typedef __attribute__((address_space(3))) v4i  sv4i_t;
#else
#define USE_ASYNC 0
#endif

// ---------------------------------------------------------------------------
// f32 -> f16 conversion (grid-stride, row-major preserved)
// ---------------------------------------------------------------------------
__global__ void cvt_f16_kernel(const float* __restrict__ in,
                               _Float16* __restrict__ out, long n) {
    long i = (long)blockIdx.x * blockDim.x + threadIdx.x;
    long stride = (long)gridDim.x * blockDim.x;
    for (; i < n; i += stride) out[i] = (_Float16)in[i];
}

// ---------------------------------------------------------------------------
// f32 [K,N] -> f16 [N,K] tiled transpose (weights become fragment-friendly:
// every GEMM tile row is then a contiguous 64B run for both A and B).
// ---------------------------------------------------------------------------
__global__ __launch_bounds__(256) void cvtT_kernel(const float* __restrict__ in,
                                                   _Float16* __restrict__ out,
                                                   int K, int N) {
    __shared__ _Float16 tile[32][33];
    const int kt = blockIdx.y * 32, nt = blockIdx.x * 32;
    const int tx = threadIdx.x & 31, ty = threadIdx.x >> 5;  // ty: 0..7
    #pragma unroll
    for (int j = 0; j < 4; ++j) {
        int k = kt + ty + j * 8;
        tile[ty + j * 8][tx] = (_Float16)in[(long)k * N + nt + tx];
    }
    __syncthreads();
    #pragma unroll
    for (int j = 0; j < 4; ++j) {
        int n = nt + ty + j * 8;
        out[(long)n * K + kt + tx] = tile[tx][ty + j * 8];
    }
}

// ---------------------------------------------------------------------------
// Generic WMMA GEMM:  out = A[rows,K] @ W[K,Ncols] + bias  (W passed PRE-
// TRANSPOSED as Wt[Ncols,K]).  Epilogues:
//   mode 0: outF = v                      (optional zero-token row shift)
//   mode 1: outF = tanh(v)+1
//   mode 2: outH = f16(gelu_exact(v))
//   mode 3: outF = v + res1 + res2        (final fused residual)
// Block tile 128x64, K-step 32; 8 waves, each 32x32 via 2x2 16x16x32 WMMA.
// Staging: async b128 global->LDS with ping-pong double buffering when
// available, else vectorized uint4 copies.
// ---------------------------------------------------------------------------
__global__ __launch_bounds__(256) void gemm_wmma_kernel(
    const _Float16* __restrict__ A, const _Float16* __restrict__ Wt,
    const float* __restrict__ bias,
    float* __restrict__ outF, _Float16* __restrict__ outH,
    const float* __restrict__ res1, const float* __restrict__ res2,
    int Nrows, int K, int Ncols, int mode, int zshift, int Lseq)
{
#if USE_ASYNC
    __shared__ alignas(16) _Float16 As[2][128 * 32];
    __shared__ alignas(16) _Float16 Bs[2][64 * 32];
#else
    __shared__ alignas(16) _Float16 As[1][128 * 32];
    __shared__ alignas(16) _Float16 Bs[1][64 * 32];
#endif
    const int tid  = threadIdx.x;
    const int col0 = blockIdx.x * 64;
    const int row0 = blockIdx.y * 128;
    const int w    = tid >> 5;
    const int ln   = tid & 31;
    const int wr   = w >> 1;            // 0..3  (row group of 32)
    const int wc   = w & 1;             // 0..1  (col group of 32)

    v8f acc[2][2] = {};
    const int nk = K / 32;

    // fragment build + 4 WMMA for one staged tile pair
    auto compute = [&](const _Float16* Asb, const _Float16* Bsb) {
        Frag16 af[2], bf[2];
        const int mA  = ln & 15;
        const int kbA = (ln < 16) ? 0 : 8;
        #pragma unroll
        for (int i = 0; i < 2; ++i) {
            const _Float16* ap = &Asb[(wr * 32 + i * 16 + mA) * 32];
            #pragma unroll
            for (int j = 0; j < 8; ++j) {
                int kk = kbA + 2 * j + ((j >= 4) ? 8 : 0);
                af[i].u32[j] = *(const unsigned int*)(ap + kk);
            }
        }
        const int nB  = ln & 15;
        const int kbB = (ln < 16) ? 0 : 8;
        #pragma unroll
        for (int i = 0; i < 2; ++i) {
            const _Float16* bp = &Bsb[(wc * 32 + i * 16 + nB) * 32 + kbB];
            #pragma unroll
            for (int j = 0; j < 8; ++j)
                bf[i].u32[j] = *(const unsigned int*)(bp + 2 * j + ((j >= 4) ? 8 : 0));
        }
        #pragma unroll
        for (int i = 0; i < 2; ++i)
            #pragma unroll
            for (int j = 0; j < 2; ++j)
                acc[i][j] = __builtin_amdgcn_wmma_f32_16x16x32_f16(
                    false, af[i].h, false, bf[j].h,
                    (short)0, acc[i][j], false, false);
    };

#if USE_ASYNC
    // issue async b128 copies of tile `it` into buffer `bi`
    auto issue = [&](int it, int bi) {
        const int k0 = it * 32;
        #pragma unroll
        for (int i = 0; i < 2; ++i) {          // A: 128x32 f16 = 512 x 16B
            int e = tid + i * 256;
            int m = e >> 2, c = e & 3;
            int gr = row0 + m;
            if (gr >= Nrows) gr = Nrows - 1;   // clamp; masked at store
            __builtin_amdgcn_global_load_async_to_lds_b128(
                (v4i*)(void*)(A + (long)gr * K + k0 + c * 8),
                (sv4i_t*)(svoid_t*)(void*)(&As[bi][m * 32 + c * 8]), 0, 0);
        }
        {                                       // B: 64x32 f16 = 256 x 16B
            int m = tid >> 2, c = tid & 3;
            __builtin_amdgcn_global_load_async_to_lds_b128(
                (v4i*)(void*)(Wt + (long)(col0 + m) * K + k0 + c * 8),
                (sv4i_t*)(svoid_t*)(void*)(&Bs[bi][m * 32 + c * 8]), 0, 0);
        }
    };
    issue(0, 0);
    for (int it = 0; it < nk; ++it) {
        const int cur = it & 1;
        __builtin_amdgcn_s_wait_asynccnt(0);   // my writes into `cur` landed
        __syncthreads();                       // everyone's landed; prior reads done
        if (it + 1 < nk) issue(it + 1, cur ^ 1);
        compute(As[cur], Bs[cur]);
        __syncthreads();                       // reads of `cur` done before rewrite
    }
#else
    for (int it = 0; it < nk; ++it) {
        const int k0 = it * 32;
        #pragma unroll
        for (int i = 0; i < 2; ++i) {
            int e = tid + i * 256;
            int m = e >> 2, c = e & 3;
            int gr = row0 + m;
            if (gr >= Nrows) gr = Nrows - 1;
            *(uint4*)&As[0][m * 32 + c * 8] =
                *(const uint4*)&A[(long)gr * K + k0 + c * 8];
        }
        {
            int m = tid >> 2, c = tid & 3;
            *(uint4*)&Bs[0][m * 32 + c * 8] =
                *(const uint4*)&Wt[(long)(col0 + m) * K + k0 + c * 8];
        }
        if (k0 + 32 < K) {
            int gr = row0 + (tid >> 1);
            if (gr < Nrows) __builtin_prefetch(&A[(long)gr * K + k0 + 32], 0, 1);
        }
        __syncthreads();
        compute(As[0], Bs[0]);
        __syncthreads();
    }
#endif

    // ---- epilogue: C/D layout lane ln, vgpr r -> (m = r + (ln<16?0:8)) ----
    #pragma unroll
    for (int i = 0; i < 2; ++i) {
        #pragma unroll
        for (int j = 0; j < 2; ++j) {
            int gcol = col0 + wc * 32 + j * 16 + (ln & 15);
            float bv = bias ? bias[gcol] : 0.f;
            #pragma unroll
            for (int r = 0; r < 8; ++r) {
                int grow = row0 + wr * 32 + i * 16 + r + ((ln < 16) ? 0 : 8);
                if (grow >= Nrows) continue;
                float v = acc[i][j][r] + bv;
                long orow = grow;
                if (zshift) {
                    int bb = grow / Lseq, l = grow % Lseq;
                    orow = (long)bb * (Lseq + 1) + l + 1;
                }
                long idx = orow * (long)Ncols + gcol;
                if (mode == 0)      outF[idx] = v;
                else if (mode == 1) outF[idx] = tanhf(v) + 1.f;
                else if (mode == 2) outH[idx] =
                    (_Float16)(0.5f * v * (1.f + erff(v * 0.70710678118654752f)));
                else                outF[idx] = v + res1[idx] + res2[idx];
            }
        }
    }
}

// ---------------------------------------------------------------------------
// Row LayerNorm over E=1024; optional zero-token rows (row % ztokP == 0 -> 0),
// optional f16 shadow output. Safe in-place.
// ---------------------------------------------------------------------------
__global__ __launch_bounds__(256) void ln_kernel(
    const float* __restrict__ in, const float* __restrict__ g,
    const float* __restrict__ b, float* __restrict__ outF,
    _Float16* __restrict__ outH, int E, int ztokP)
{
    const int row = blockIdx.x;
    const long base = (long)row * E;
    const int t = threadIdx.x;
    __shared__ float red[256];

    if (ztokP && (row % ztokP == 0)) {
        for (int c = t; c < E; c += 256) {
            outF[base + c] = 0.f;
            if (outH) outH[base + c] = (_Float16)0.f;
        }
        return;
    }
    float s = 0.f;
    for (int c = t; c < E; c += 256) s += in[base + c];
    red[t] = s; __syncthreads();
    for (int o = 128; o > 0; o >>= 1) { if (t < o) red[t] += red[t + o]; __syncthreads(); }
    float mu = red[0] / (float)E;
    __syncthreads();
    float v = 0.f;
    for (int c = t; c < E; c += 256) { float d = in[base + c] - mu; v += d * d; }
    red[t] = v; __syncthreads();
    for (int o = 128; o > 0; o >>= 1) { if (t < o) red[t] += red[t + o]; __syncthreads(); }
    float rs = rsqrtf(red[0] / (float)E + 1e-5f);
    __syncthreads();
    for (int c = t; c < E; c += 256) {
        float y = (in[base + c] - mu) * rs * g[c] + b[c];
        outF[base + c] = y;
        if (outH) outH[base + c] = (_Float16)y;
    }
}

// ---------------------------------------------------------------------------
// RALA: per (b,h): qg = mean_l(Qh)/sqrt(D); scores = qg.Kz over Lk=L+1;
// alpha = softmax(scores)*L. One block per (b,h). Scores staged in LDS.
// ---------------------------------------------------------------------------
__global__ __launch_bounds__(256) void rala_kernel(
    const float* __restrict__ Q, const float* __restrict__ Kz,
    float* __restrict__ alpha, int Bn, int L, int H, int E)
{
    const int bh = blockIdx.x, b = bh / H, h = bh % H;
    const int Lk = L + 1;
    __shared__ float qg[64];
    __shared__ float sc[Lc + 1];
    __shared__ float red[256];
    const int t = threadIdx.x;

    const int d = t & 63, grp = t >> 6;
    float s = 0.f;
    for (int l = grp; l < L; l += 4)
        s += Q[((long)b * L + l) * E + h * 64 + d];
    red[t] = s; __syncthreads();
    if (t < 64)
        qg[t] = (red[t] + red[64 + t] + red[128 + t] + red[192 + t])
                / (float)L * rsqrtf((float)Dc);
    __syncthreads();

    float lmax = -1e30f;
    for (int l = t; l < Lk; l += 256) {
        const float* kp = &Kz[((long)b * Lk + l) * E + h * 64];
        float sv = 0.f;
        #pragma unroll 8
        for (int dd = 0; dd < 64; ++dd) sv += qg[dd] * kp[dd];
        sc[l] = sv;
        lmax = fmaxf(lmax, sv);
    }
    red[t] = lmax; __syncthreads();
    for (int o = 128; o > 0; o >>= 1) { if (t < o) red[t] = fmaxf(red[t], red[t + o]); __syncthreads(); }
    float M = red[0]; __syncthreads();
    float ls = 0.f;
    for (int l = t; l < Lk; l += 256) ls += expf(sc[l] - M);
    red[t] = ls; __syncthreads();
    for (int o = 128; o > 0; o >>= 1) { if (t < o) red[t] += red[t + o]; __syncthreads(); }
    float inv = (float)L / red[0];
    __syncthreads();
    for (int l = t; l < Lk; l += 256)
        alpha[(long)bh * Lk + l] = expf(sc[l] - M) * inv;
}

// ---------------------------------------------------------------------------
// Per-head KV = sum_l (pK*alpha)(d) * V(e)  [64x64], K_sum = sum_l pK*alpha.
// ---------------------------------------------------------------------------
__global__ __launch_bounds__(256) void kv_kernel(
    const float* __restrict__ pK, const float* __restrict__ Vz,
    const float* __restrict__ alpha, float* __restrict__ KV,
    float* __restrict__ Ksum, int Bn, int L, int H, int E)
{
    const int bh = blockIdx.x, b = bh / H, h = bh % H;
    const int Lk = L + 1;
    __shared__ float sk[8][64];
    __shared__ float sv[8][64];
    __shared__ float sa[8];
    const int t = threadIdx.x;
    const int e = t & 63, dg = t >> 6;
    float acc[16];
    #pragma unroll
    for (int i = 0; i < 16; ++i) acc[i] = 0.f;
    float ks = 0.f;

    for (int l0 = 0; l0 < Lk; l0 += 8) {
        int nl = min(8, Lk - l0);
        for (int i = t; i < nl * 64; i += 256) {
            int li = i >> 6, c = i & 63;
            long rb = ((long)b * Lk + (l0 + li)) * E + h * 64;
            sk[li][c] = pK[rb + c];
            sv[li][c] = Vz[rb + c];
        }
        if (t < nl) sa[t] = alpha[(long)bh * Lk + l0 + t];
        __syncthreads();
        for (int li = 0; li < nl; ++li) {
            float av = sa[li];
            float ve = sv[li][e] * av;
            #pragma unroll
            for (int i = 0; i < 16; ++i) acc[i] += sk[li][dg * 16 + i] * ve;
            if (t < 64) ks += sk[li][t] * av;
        }
        __syncthreads();
    }
    long ob = (long)bh * 64 * 64;
    #pragma unroll
    for (int i = 0; i < 16; ++i) KV[ob + (long)(dg * 16 + i) * 64 + e] = acc[i];
    if (t < 64) Ksum[(long)bh * 64 + t] = ks;
}

// ---------------------------------------------------------------------------
// attn = (pQ @ KV) / (pQ.Ksum + eps)  per head; KV cached in LDS.
// ---------------------------------------------------------------------------
__global__ __launch_bounds__(256) void num_kernel(
    const float* __restrict__ pQ, const float* __restrict__ KV,
    const float* __restrict__ Ksum, float* __restrict__ attn,
    int Bn, int L, int H, int E)
{
    const int bh = blockIdx.y, b = bh / H, h = bh % H;
    const int l0 = blockIdx.x * 32;
    __shared__ float kvs[64][64];
    __shared__ float qs[32][64];
    __shared__ float ksm[64];
    __shared__ float den[32];
    const int t = threadIdx.x;

    for (int i = t; i < 4096; i += 256) kvs[i >> 6][i & 63] = KV[(long)bh * 4096 + i];
    if (t < 64) ksm[t] = Ksum[(long)bh * 64 + t];
    for (int i = t; i < 2048; i += 256) {
        int li = i >> 6, c = i & 63;
        qs[li][c] = pQ[((long)b * L + l0 + li) * E + h * 64 + c];
    }
    __syncthreads();
    if (t < 32) {
        float s = 0.f;
        #pragma unroll 8
        for (int dd = 0; dd < 64; ++dd) s += qs[t][dd] * ksm[dd];
        den[t] = s + 1e-6f;
    }
    __syncthreads();
    const int li = t >> 3, e0 = (t & 7) * 8;
    float o[8] = {0, 0, 0, 0, 0, 0, 0, 0};
    for (int dd = 0; dd < 64; ++dd) {
        float q = qs[li][dd];
        #pragma unroll
        for (int j = 0; j < 8; ++j) o[j] += q * kvs[dd][e0 + j];
    }
    float dn = den[li];
    long rb = ((long)b * L + l0 + li) * E + h * 64;
    #pragma unroll
    for (int j = 0; j < 8; ++j) attn[rb + e0 + j] = o[j] / dn;
}

// ---------------------------------------------------------------------------
// Gated grouped conv (k=5, pad=2, groups=H): out = attn + gamma*(conv+cb)
// ---------------------------------------------------------------------------
__global__ __launch_bounds__(256) void conv_kernel(
    const float* __restrict__ attn, const float* __restrict__ cw,
    const float* __restrict__ cb, const float* __restrict__ gamma,
    float* __restrict__ out, int Bn, int L, int H, int E)
{
    const int bh = blockIdx.y, b = bh / H, h = bh % H;
    const int l0 = blockIdx.x * 32;
    __shared__ float xs[36][64];
    const int t = threadIdx.x;
    for (int i = t; i < 36 * 64; i += 256) {
        int li = i >> 6, c = i & 63;
        int l = l0 + li - 2;
        xs[li][c] = (l >= 0 && l < L) ? attn[((long)b * L + l) * E + h * 64 + c] : 0.f;
    }
    __syncthreads();
    const int co = t & 63, ls0 = (t >> 6) * 8;
    float acc[8] = {0, 0, 0, 0, 0, 0, 0, 0};
    const float* wp = &cw[(long)(h * 64 + co) * 64 * 5];
    for (int ci = 0; ci < 64; ++ci) {
        #pragma unroll
        for (int k = 0; k < 5; ++k) {
            float w = wp[ci * 5 + k];
            #pragma unroll
            for (int j = 0; j < 8; ++j) acc[j] += xs[ls0 + j + k][ci] * w;
        }
    }
    float g = gamma[0];
    float cbv = cb[h * 64 + co];
    #pragma unroll
    for (int j = 0; j < 8; ++j) {
        long idx = ((long)b * L + l0 + ls0 + j) * E + h * 64 + co;
        out[idx] = attn[idx] + g * (acc[j] + cbv);
    }
}

// ---------------------------------------------------------------------------
// Host-side orchestration
// ---------------------------------------------------------------------------
extern "C" void kernel_launch(void* const* d_in, const int* in_sizes, int n_in,
                              void* d_out, int out_size, void* d_ws, size_t ws_size,
                              hipStream_t stream)
{
    const float* x      = (const float*)d_in[0];
    const float* Wq     = (const float*)d_in[1];
    const float* bq     = (const float*)d_in[2];
    const float* lnq_g  = (const float*)d_in[3];
    const float* lnq_b  = (const float*)d_in[4];
    const float* Wk     = (const float*)d_in[5];
    const float* bk     = (const float*)d_in[6];
    const float* lnk_g  = (const float*)d_in[7];
    const float* lnk_b  = (const float*)d_in[8];
    const float* Wv     = (const float*)d_in[9];
    const float* bv     = (const float*)d_in[10];
    const float* lnv_g  = (const float*)d_in[11];
    const float* lnv_b  = (const float*)d_in[12];
    const float* Wphi_q = (const float*)d_in[13];
    const float* bphi_q = (const float*)d_in[14];
    const float* Wphi_k = (const float*)d_in[15];
    const float* bphi_k = (const float*)d_in[16];
    const float* lna_g  = (const float*)d_in[17];
    const float* lna_b  = (const float*)d_in[18];
    const float* W1     = (const float*)d_in[19];
    const float* b1     = (const float*)d_in[20];
    const float* W2     = (const float*)d_in[21];
    const float* b2     = (const float*)d_in[22];
    const float* gamma  = (const float*)d_in[23];
    const float* conv_w = (const float*)d_in[24];
    const float* conv_b = (const float*)d_in[25];

    // ---- workspace carve-up ----
    char* p = (char*)d_ws;
    auto alloc = [&](size_t bytes) -> void* {
        void* r = (void*)p;
        p += (bytes + 255) & ~(size_t)255;
        return r;
    };
    _Float16* xb    = (_Float16*)alloc(Nc * Ec * 2);
    _Float16* WqT   = (_Float16*)alloc((size_t)Ec * Ec * 2);    // [n][k]
    _Float16* WkT   = (_Float16*)alloc((size_t)Ec * Ec * 2);
    _Float16* WvT   = (_Float16*)alloc((size_t)Ec * Ec * 2);
    _Float16* WpqT  = (_Float16*)alloc((size_t)Ec * Ec * 2);
    _Float16* WpkT  = (_Float16*)alloc((size_t)Ec * Ec * 2);
    _Float16* W1T   = (_Float16*)alloc((size_t)Ec * MLPc * 2);  // [MLP][E]
    _Float16* W2T   = (_Float16*)alloc((size_t)MLPc * Ec * 2);  // [E][MLP]
    float*    Qf    = (float*)alloc(Nc * Ec * 4);
    _Float16* Qh    = (_Float16*)alloc(Nc * Ec * 2);
    float*    Kzf   = (float*)alloc(Nzc * Ec * 4);
    _Float16* Kzh   = (_Float16*)alloc(Nzc * Ec * 2);
    float*    Vzf   = (float*)alloc(Nzc * Ec * 4);
    float*    pQ    = (float*)alloc(Nc * Ec * 4);
    float*    pK    = (float*)alloc(Nzc * Ec * 4);
    float*    alphaB = (float*)alloc((size_t)Bc * Hc * (Lc + 1) * 4);
    float*    KV    = (float*)alloc((size_t)Bc * Hc * 64 * 64 * 4);
    float*    Ksum  = (float*)alloc((size_t)Bc * Hc * 64 * 4);
    float*    attnR = (float*)alloc(Nc * Ec * 4);
    float*    attnC = (float*)alloc(Nc * Ec * 4);
    float*    attnL = (float*)alloc(Nc * Ec * 4);
    _Float16* attnLh = (_Float16*)alloc(Nc * Ec * 2);
    _Float16* Hmid  = (_Float16*)alloc(Nc * MLPc * 2);

    // ---- conversions: activations row-major, weights transposed ----
    {
        long n = Nc * Ec;
        cvt_f16_kernel<<<4096, 256, 0, stream>>>(x, xb, n);
    }
    auto cvtT = [&](const float* src, _Float16* dst, int K, int N) {
        dim3 grid(N / 32, K / 32);
        cvtT_kernel<<<grid, 256, 0, stream>>>(src, dst, K, N);
    };
    cvtT(Wq,     WqT,  Ec, Ec);
    cvtT(Wk,     WkT,  Ec, Ec);
    cvtT(Wv,     WvT,  Ec, Ec);
    cvtT(Wphi_q, WpqT, Ec, Ec);
    cvtT(Wphi_k, WpkT, Ec, Ec);
    cvtT(W1,     W1T,  Ec, MLPc);
    cvtT(W2,     W2T,  MLPc, Ec);

    auto gemm = [&](const _Float16* A, const _Float16* Wt, const float* bias,
                    float* oF, _Float16* oH, const float* r1, const float* r2,
                    long rows, int K, int Ncols, int mode, int zshift) {
        dim3 grid(Ncols / 64, (unsigned)((rows + 127) / 128));
        gemm_wmma_kernel<<<grid, 256, 0, stream>>>(
            A, Wt, bias, oF, oH, r1, r2, (int)rows, K, Ncols, mode, zshift, Lc);
    };

    // ---- QKV projections (K/V row-shifted past zero token) ----
    gemm(xb, WqT, bq, Qf,  nullptr, nullptr, nullptr, Nc, Ec, Ec, 0, 0);
    gemm(xb, WkT, bk, Kzf, nullptr, nullptr, nullptr, Nc, Ec, Ec, 0, 1);
    gemm(xb, WvT, bv, Vzf, nullptr, nullptr, nullptr, Nc, Ec, Ec, 0, 1);

    // ---- LayerNorms (zero-token rows forced to 0 for K/V) ----
    ln_kernel<<<(unsigned)Nc,  256, 0, stream>>>(Qf,  lnq_g, lnq_b, Qf,  Qh,      Ec, 0);
    ln_kernel<<<(unsigned)Nzc, 256, 0, stream>>>(Kzf, lnk_g, lnk_b, Kzf, Kzh,     Ec, Lc + 1);
    ln_kernel<<<(unsigned)Nzc, 256, 0, stream>>>(Vzf, lnv_g, lnv_b, Vzf, nullptr, Ec, Lc + 1);

    // ---- phi feature maps: tanh(.)+1 ----
    gemm(Qh,  WpqT, bphi_q, pQ, nullptr, nullptr, nullptr, Nc,  Ec, Ec, 1, 0);
    gemm(Kzh, WpkT, bphi_k, pK, nullptr, nullptr, nullptr, Nzc, Ec, Ec, 1, 0);

    // ---- RALA rescaling: alpha[b,h,l] ----
    rala_kernel<<<Bc * Hc, 256, 0, stream>>>(Qf, Kzf, alphaB, Bc, Lc, Hc, Ec);

    // ---- per-head KV and K_sum ----
    kv_kernel<<<Bc * Hc, 256, 0, stream>>>(pK, Vzf, alphaB, KV, Ksum, Bc, Lc, Hc, Ec);

    // ---- attn = (pQ@KV)/(pQ.Ksum+eps) ----
    num_kernel<<<dim3(Lc / 32, Bc * Hc), 256, 0, stream>>>(
        pQ, KV, Ksum, attnR, Bc, Lc, Hc, Ec);

    // ---- grouped local conv + gate ----
    conv_kernel<<<dim3(Lc / 32, Bc * Hc), 256, 0, stream>>>(
        attnR, conv_w, conv_b, gamma, attnC, Bc, Lc, Hc, Ec);

    // ---- attention LayerNorm (f32 + f16 shadow for MLP) ----
    ln_kernel<<<(unsigned)Nc, 256, 0, stream>>>(attnC, lna_g, lna_b, attnL, attnLh, Ec, 0);

    // ---- MLP: gelu(attnL@W1+b1)@W2+b2, fused final residual out = h+attnL+x ----
    gemm(attnLh, W1T, b1, nullptr, Hmid, nullptr, nullptr, Nc, Ec, MLPc, 2, 0);
    gemm(Hmid,   W2T, b2, (float*)d_out, nullptr, attnL, x, Nc, MLPc, Ec, 3, 0);
}
